// IntraAttention_74363063763252
// MI455X (gfx1250) — matrix operand order repaired and checked by reading, not compiled, not run on hardware
//
#include <hip/hip_runtime.h>
#include <hip/hip_bf16.h>
#include <math.h>

// ---------------------------------------------------------------------------
// IntraAttention for MI455X (gfx1250): bf16 WMMA, f32 accumulate,
// async global->LDS double-buffered GEMM pipeline (ASYNCcnt), 64x64 wave tiles.
// ---------------------------------------------------------------------------

typedef __attribute__((ext_vector_type(16))) __bf16 bf16x16;
typedef __attribute__((ext_vector_type(8)))  float  f32x8;
typedef __attribute__((ext_vector_type(4)))  int    i32x4;

#define NTOK   4096
#define DMODEL 1024
#define APAD   40          // padded LDS row stride (elements) for 32-wide K panel

// ---- WMMA ----------------------------------------------------------------

__device__ __forceinline__ f32x8 wmma_bf16(bf16x16 a, bf16x16 b, f32x8 c) {
  return __builtin_amdgcn_wmma_f32_16x16x32_bf16(
      false, a, false, b, (short)0, c, false, false);
}

// ---- async global -> LDS (16B per lane), ASYNCcnt-tracked -----------------

__device__ __forceinline__ void async_cp16(const __bf16* g, const __bf16* l) {
#if __has_builtin(__builtin_amdgcn_global_load_async_to_lds_b128)
  __builtin_amdgcn_global_load_async_to_lds_b128(
      (__attribute__((address_space(1))) i32x4*)(unsigned long long)g,
      (__attribute__((address_space(3))) i32x4*)(unsigned)(unsigned long long)l,
      0, 0);
#else
  asm volatile("global_load_async_to_lds_b128 %0, %1, off"
               :: "v"((unsigned)(unsigned long long)l), "v"(g) : "memory");
#endif
}

template <int N>
__device__ __forceinline__ void wait_async() {
#if __has_builtin(__builtin_amdgcn_s_wait_asynccnt)
  __builtin_amdgcn_s_wait_asynccnt(N);
#else
  asm volatile("s_wait_asynccnt %0" :: "i"(N) : "memory");
#endif
}

// ---- fragment loads from LDS (32-deep K panel, padded stride APAD) --------
// A frag (16x32 MxK): lane&15 = M row; K 16*hi..16*hi+15.
__device__ __forceinline__ bf16x16 lds_a_frag(const __bf16* As, int m, int lane) {
  int r  = lane & 15;
  int hi = (lane >> 4) & 1;
  const __bf16* p = As + (m + r) * APAD + hi * 16;
  union { bf16x16 v; uint4 q[2]; } u;
  u.q[0] = *(const uint4*)(p);
  u.q[1] = *(const uint4*)(p + 8);
  return u.v;
}
// B frag (32x16 KxN) from N-major rows: lane&15 = N col; K 16*hi..16*hi+15.
__device__ __forceinline__ bf16x16 lds_b_frag(const __bf16* Bs, int n, int lane) {
  int c  = lane & 15;
  int hi = (lane >> 4) & 1;
  const __bf16* p = Bs + (n + c) * APAD + hi * 16;
  union { bf16x16 v; uint4 q[2]; } u;
  u.q[0] = *(const uint4*)(p);
  u.q[1] = *(const uint4*)(p + 8);
  return u.v;
}

// direct-from-memory frags (used by final_kernel)
__device__ __forceinline__ bf16x16 load_a_frag(const __bf16* A, int lda,
                                               int m0, int k0, int lane) {
  int r = lane & 15, hi = (lane >> 4) & 1;
  const __bf16* p = A + (size_t)(m0 + r) * lda + k0 + hi * 16;
  union { bf16x16 v; uint4 q[2]; } u;
  u.q[0] = *(const uint4*)(p);
  u.q[1] = *(const uint4*)(p + 8);
  return u.v;
}
__device__ __forceinline__ bf16x16 load_b_frag(const __bf16* B, int ldb,
                                               int n0, int k0, int lane) {
  int c = lane & 15, hi = (lane >> 4) & 1;
  const __bf16* p = B + (size_t)(n0 + c) * ldb + k0 + hi * 16;
  union { bf16x16 v; uint4 q[2]; } u;
  u.q[0] = *(const uint4*)(p);
  u.q[1] = *(const uint4*)(p + 8);
  return u.v;
}

// ---- stage one 128x32 A panel + 256x32 B panel into LDS (async) -----------
// 512 A-chunks + 1024 B-chunks of 16B; 256 threads x 6 issues.
__device__ __forceinline__ void stage_load(const __bf16* __restrict__ A, int lda,
                                           const __bf16* __restrict__ W, int ldw,
                                           int bm, int bn, int k, int tid,
                                           __bf16* As, __bf16* Bs) {
  #pragma unroll
  for (int i = 0; i < 2; ++i) {
    int c = tid + i * 256, r = c >> 2, off = (c & 3) * 8;
    async_cp16(A + (size_t)(bm + r) * lda + k + off, As + r * APAD + off);
  }
  #pragma unroll
  for (int i = 0; i < 4; ++i) {
    int c = tid + i * 256, r = c >> 2, off = (c & 3) * 8;
    async_cp16(W + (size_t)(bn + r) * ldw + k + off, Bs + r * APAD + off);
  }
}

// ---- shared GEMM mainloop: block tile 128x256, 8 waves (2x4) x 64x64 ------
__device__ __forceinline__ void gemm_mainloop(const __bf16* __restrict__ A, int lda,
                                              const __bf16* __restrict__ W, int ldw,
                                              int K, int bm, int bn,
                                              int wm, int wn, int lane, int tid,
                                              __bf16* As, __bf16* Bs,
                                              f32x8 acc[4][4]) {
  const int KB = K / 32;
  stage_load(A, lda, W, ldw, bm, bn, 0, tid, As, Bs);
  for (int kb = 0; kb < KB; ++kb) {
    int cur = kb & 1;
    __bf16* Asc = As + cur * (128 * APAD);
    __bf16* Bsc = Bs + cur * (256 * APAD);
    if (kb + 1 < KB) {
      stage_load(A, lda, W, ldw, bm, bn, (kb + 1) * 32, tid,
                 As + (cur ^ 1) * (128 * APAD), Bs + (cur ^ 1) * (256 * APAD));
      wait_async<6>();        // previous stage complete; next 6 still in flight
    } else {
      wait_async<0>();
    }
    __syncthreads();
    bf16x16 af[4], bfr[4];
    #pragma unroll
    for (int i = 0; i < 4; ++i) af[i]  = lds_a_frag(Asc, wm * 64 + i * 16, lane);
    #pragma unroll
    for (int j = 0; j < 4; ++j) bfr[j] = lds_b_frag(Bsc, wn * 64 + j * 16, lane);
    #pragma unroll
    for (int i = 0; i < 4; ++i)
      #pragma unroll
      for (int j = 0; j < 4; ++j)
        acc[i][j] = wmma_bf16(af[i], bfr[j], acc[i][j]);
    __syncthreads();          // everyone done reading cur before it is refilled
  }
}

// ---- GEMM, bf16 out (+bias, optional transposed store) --------------------
__global__ __launch_bounds__(256) void gemm_lds_bf16out(
    const __bf16* __restrict__ A, int lda,
    const __bf16* __restrict__ W, int ldw,
    const float* __restrict__ bias,
    __bf16* __restrict__ D, int ldd,
    int K, int transposed_store)
{
  __shared__ __bf16 As[2 * 128 * APAD];
  __shared__ __bf16 Bs[2 * 256 * APAD];
  int tid  = threadIdx.x;
  int lane = tid & 31;
  int wave = tid >> 5;
  int wm = wave >> 2, wn = wave & 3;
  int bm = blockIdx.x * 128, bn = blockIdx.y * 256;

  f32x8 acc[4][4] = {};
  gemm_mainloop(A, lda, W, ldw, K, bm, bn, wm, wn, lane, tid, As, Bs, acc);

  int nl = lane & 15, mb = (lane >> 4) * 8;
  for (int ti = 0; ti < 4; ++ti)
    for (int tj = 0; tj < 4; ++tj) {
      int n = bn + wn * 64 + tj * 16 + nl;
      float bv = bias ? bias[n] : 0.0f;
      #pragma unroll
      for (int j = 0; j < 8; ++j) {
        int m = bm + wm * 64 + ti * 16 + mb + j;
        float v = acc[ti][tj][j] + bv;
        if (transposed_store) D[(size_t)n * ldd + m] = (__bf16)v;
        else                  D[(size_t)m * ldd + n] = (__bf16)v;
      }
    }
}

// ---- GEMM, fp32 out with scale (scores) -----------------------------------
__global__ __launch_bounds__(256) void gemm_lds_f32out(
    const __bf16* __restrict__ A, int lda,
    const __bf16* __restrict__ W, int ldw,
    float* __restrict__ S, int lds_, float scale, int K)
{
  __shared__ __bf16 As[2 * 128 * APAD];
  __shared__ __bf16 Bs[2 * 256 * APAD];
  int tid  = threadIdx.x;
  int lane = tid & 31;
  int wave = tid >> 5;
  int wm = wave >> 2, wn = wave & 3;
  int bm = blockIdx.x * 128, bn = blockIdx.y * 256;

  f32x8 acc[4][4] = {};
  gemm_mainloop(A, lda, W, ldw, K, bm, bn, wm, wn, lane, tid, As, Bs, acc);

  int nl = lane & 15, mb = (lane >> 4) * 8;
  for (int ti = 0; ti < 4; ++ti)
    for (int tj = 0; tj < 4; ++tj) {
      int n = bn + wn * 64 + tj * 16 + nl;
      #pragma unroll
      for (int j = 0; j < 8; ++j) {
        int m = bm + wm * 64 + ti * 16 + mb + j;
        S[(size_t)m * lds_ + n] = acc[ti][tj][j] * scale;
      }
    }
}

// ---- fp32 -> bf16 conversion ---------------------------------------------
__global__ __launch_bounds__(256) void cvt_f32_bf16(const float* __restrict__ in,
                                                    __bf16* __restrict__ out,
                                                    int n) {
  int i = (blockIdx.x * 256 + threadIdx.x) * 4;
  if (i < n) {
    float4 v = *(const float4*)(in + i);
    out[i + 0] = (__bf16)v.x;
    out[i + 1] = (__bf16)v.y;
    out[i + 2] = (__bf16)v.z;
    out[i + 3] = (__bf16)v.w;
  }
}

// ---- row softmax, fp32 in -> bf16 alpha written in place ------------------
__global__ __launch_bounds__(256) void softmax_kernel(float* __restrict__ S) {
  int row = blockIdx.x;
  int t   = threadIdx.x;
  float*  srow = S + (size_t)row * NTOK;
  __bf16* arow = (__bf16*)srow;

  float v[NTOK / 256];
  #pragma unroll
  for (int i = 0; i < NTOK / 256; ++i) v[i] = srow[t + 256 * i];

  __shared__ float red[256];
  float mx = -INFINITY;
  #pragma unroll
  for (int i = 0; i < NTOK / 256; ++i) mx = fmaxf(mx, v[i]);
  red[t] = mx; __syncthreads();
  for (int s = 128; s > 0; s >>= 1) {
    if (t < s) red[t] = fmaxf(red[t], red[t + s]);
    __syncthreads();
  }
  float rmax = red[0]; __syncthreads();

  float sum = 0.0f;
  #pragma unroll
  for (int i = 0; i < NTOK / 256; ++i) { v[i] = expf(v[i] - rmax); sum += v[i]; }
  red[t] = sum; __syncthreads();
  for (int s = 128; s > 0; s >>= 1) {
    if (t < s) red[t] += red[t + s];
    __syncthreads();
  }
  float inv = 1.0f / red[0]; __syncthreads();

  #pragma unroll
  for (int i = 0; i < NTOK / 256; ++i)
    arow[t + 256 * i] = (__bf16)(v[i] * inv);
}

// ---- final: out = (x*Wl^T + bl) * (a * sigmoid(b)), h = x*Wa^T + ba -------
__global__ __launch_bounds__(256) void final_kernel(
    const __bf16* __restrict__ Xc,   // 4096 x 2048 = [V', Q]
    const __bf16* __restrict__ Wlb,  // 1024 x 2048
    const __bf16* __restrict__ Wab,  // 2048 x 2048
    const float* __restrict__ bl,
    const float* __restrict__ ba,
    float* __restrict__ out)         // 4096 x 1024
{
  int lane = threadIdx.x & 31;
  int wave = threadIdx.x >> 5;
  int m0 = blockIdx.x * 64  + (wave >> 2) * 32;
  int n0 = blockIdx.y * 128 + (wave & 3) * 32;

  f32x8 accl[2][2] = {}, acca[2][2] = {}, accg[2][2] = {};
  for (int k = 0; k < 2 * DMODEL; k += 32) {
    if (k + 128 < 2 * DMODEL) {
      __builtin_prefetch(Xc  + (size_t)(m0 + (lane & 15)) * 2048 + k + 128);
      __builtin_prefetch(Wab + (size_t)(n0 + (lane & 15)) * 2048 + k + 128);
    }
    bf16x16 a0 = load_a_frag(Xc, 2048, m0,      k, lane);
    bf16x16 a1 = load_a_frag(Xc, 2048, m0 + 16, k, lane);
    bf16x16 l0 = load_b_frag(Wlb, 2048, n0,          k, lane);
    bf16x16 l1 = load_b_frag(Wlb, 2048, n0 + 16,     k, lane);
    bf16x16 p0 = load_b_frag(Wab, 2048, n0,          k, lane);
    bf16x16 p1 = load_b_frag(Wab, 2048, n0 + 16,     k, lane);
    bf16x16 g0 = load_b_frag(Wab, 2048, n0 + DMODEL,      k, lane);
    bf16x16 g1 = load_b_frag(Wab, 2048, n0 + DMODEL + 16, k, lane);
    accl[0][0] = wmma_bf16(a0, l0, accl[0][0]);
    accl[0][1] = wmma_bf16(a0, l1, accl[0][1]);
    accl[1][0] = wmma_bf16(a1, l0, accl[1][0]);
    accl[1][1] = wmma_bf16(a1, l1, accl[1][1]);
    acca[0][0] = wmma_bf16(a0, p0, acca[0][0]);
    acca[0][1] = wmma_bf16(a0, p1, acca[0][1]);
    acca[1][0] = wmma_bf16(a1, p0, acca[1][0]);
    acca[1][1] = wmma_bf16(a1, p1, acca[1][1]);
    accg[0][0] = wmma_bf16(a0, g0, accg[0][0]);
    accg[0][1] = wmma_bf16(a0, g1, accg[0][1]);
    accg[1][0] = wmma_bf16(a1, g0, accg[1][0]);
    accg[1][1] = wmma_bf16(a1, g1, accg[1][1]);
  }

  int nl = lane & 15, mb = (lane >> 4) * 8;
  for (int ti = 0; ti < 2; ++ti)
    for (int tj = 0; tj < 2; ++tj) {
      int n = n0 + tj * 16 + nl;
      float bvl = bl[n], bva = ba[n], bvg = ba[n + DMODEL];
      #pragma unroll
      for (int j = 0; j < 8; ++j) {
        int m = m0 + ti * 16 + mb + j;
        float x1 = accl[ti][tj][j] + bvl;
        float av = acca[ti][tj][j] + bva;
        float gv = accg[ti][tj][j] + bvg;
        float sig = 1.0f / (1.0f + expf(-gv));
        out[(size_t)m * DMODEL + n] = x1 * (av * sig);
      }
    }
}

// ---------------------------------------------------------------------------

extern "C" void kernel_launch(void* const* d_in, const int* in_sizes, int n_in,
                              void* d_out, int out_size, void* d_ws, size_t ws_size,
                              hipStream_t stream) {
  const float* X  = (const float*)d_in[0];
  const float* Wq = (const float*)d_in[1];
  const float* bq = (const float*)d_in[2];
  const float* Wk = (const float*)d_in[3];
  const float* bk = (const float*)d_in[4];
  const float* Wv = (const float*)d_in[5];
  const float* bv = (const float*)d_in[6];
  const float* Wl = (const float*)d_in[7];
  const float* bl = (const float*)d_in[8];
  const float* Wa = (const float*)d_in[9];
  const float* ba = (const float*)d_in[10];
  float* out = (float*)d_out;

  char* ws = (char*)d_ws;
  const size_t MB = 1024 * 1024;
  __bf16* Xb  = (__bf16*)(ws + 0 * MB);    // 4096x1024          (8 MB)
  __bf16* Kbf = (__bf16*)(ws + 8 * MB);    // 4096x1024          (8 MB)
  __bf16* Vt  = (__bf16*)(ws + 16 * MB);   // 1024x4096 (V^T)    (8 MB)
  __bf16* Xc  = (__bf16*)(ws + 24 * MB);   // 4096x2048 [V',Q]   (16 MB)
  __bf16* Wqb = (__bf16*)(ws + 40 * MB);   // 1024x1024          (2 MB)
  __bf16* Wkb = (__bf16*)(ws + 42 * MB);
  __bf16* Wvb = (__bf16*)(ws + 44 * MB);
  __bf16* Wlb = (__bf16*)(ws + 46 * MB);   // 1024x2048          (4 MB)
  __bf16* Wab = (__bf16*)(ws + 50 * MB);   // 2048x2048          (8 MB)
  float*  S   = (float*) (ws + 58 * MB);   // 4096x4096 f32      (64 MB)

  cvt_f32_bf16<<<(NTOK * DMODEL) / 1024, 256, 0, stream>>>(X,  Xb,  NTOK * DMODEL);
  cvt_f32_bf16<<<(DMODEL * DMODEL) / 1024, 256, 0, stream>>>(Wq, Wqb, DMODEL * DMODEL);
  cvt_f32_bf16<<<(DMODEL * DMODEL) / 1024, 256, 0, stream>>>(Wk, Wkb, DMODEL * DMODEL);
  cvt_f32_bf16<<<(DMODEL * DMODEL) / 1024, 256, 0, stream>>>(Wv, Wvb, DMODEL * DMODEL);
  cvt_f32_bf16<<<(DMODEL * 2 * DMODEL) / 1024, 256, 0, stream>>>(Wl, Wlb, DMODEL * 2 * DMODEL);
  cvt_f32_bf16<<<(2 * DMODEL * 2 * DMODEL) / 1024, 256, 0, stream>>>(Wa, Wab, 4 * DMODEL * DMODEL);

  dim3 blk(256);
  dim3 gProj(NTOK / 128, DMODEL / 256);    // 32 x 4

  // Q -> right half of Xc (implicit concat), K -> Kbf, V -> Vt (transposed)
  gemm_lds_bf16out<<<gProj, blk, 0, stream>>>(Xb, DMODEL, Wqb, DMODEL, bq,
                                              Xc + DMODEL, 2 * DMODEL, DMODEL, 0);
  gemm_lds_bf16out<<<gProj, blk, 0, stream>>>(Xb, DMODEL, Wkb, DMODEL, bk,
                                              Kbf, DMODEL, DMODEL, 0);
  gemm_lds_bf16out<<<gProj, blk, 0, stream>>>(Xb, DMODEL, Wvb, DMODEL, bv,
                                              Vt, NTOK, DMODEL, 1);

  // scores = Q K^T / sqrt(d)
  dim3 gScore(NTOK / 128, NTOK / 256);     // 32 x 16
  gemm_lds_f32out<<<gScore, blk, 0, stream>>>(Xc + DMODEL, 2 * DMODEL,
                                              Kbf, DMODEL,
                                              S, NTOK, 1.0f / 32.0f, DMODEL);

  // softmax rows, alpha (bf16) written in place over S
  softmax_kernel<<<NTOK, blk, 0, stream>>>(S);

  // V' = alpha @ V -> left half of Xc (alpha bf16 view of S, row stride 2*NTOK)
  gemm_lds_bf16out<<<gProj, blk, 0, stream>>>((const __bf16*)S, 2 * NTOK,
                                              Vt, NTOK, nullptr,
                                              Xc, 2 * DMODEL, NTOK, 0);

  // out = (x Wl^T + bl) * (a * sigmoid(b))
  final_kernel<<<dim3(NTOK / 64, DMODEL / 128), blk, 0, stream>>>(Xc, Wlb, Wab, bl, ba, out);
}
